// HW_E_GATNet_61813169324884
// MI455X (gfx1250) — compile-verified, matrix operand order checked by reading, not run on hardware
//
#include <hip/hip_runtime.h>

// ---------------------------------------------------------------------------
// Problem constants (match reference setup_inputs)
// ---------------------------------------------------------------------------
#define NN    50000
#define EE    800000
#define D_IN  128
#define E_DIM 16
#define HIDN  64
#define DOUT  64

typedef __attribute__((ext_vector_type(16))) _Float16 v16h;
typedef __attribute__((ext_vector_type(8)))  float    v8f;

// ---------------------------------------------------------------------------
// WMMA fragment helpers (CDNA5 ISA 7.12.2, 16-bit A 16x32 layout)
//   half index i (0..15) -> K offset for this lane
// ---------------------------------------------------------------------------
__device__ __forceinline__ int frag_k(int lane, int i) {
    int v = i >> 1;
    return ((v & 4) ? 16 : 0) + 2 * (v & 3) + ((lane & 16) ? 8 : 0) + (i & 1);
}

// ordered-uint mapping for float atomic max
__device__ __forceinline__ unsigned ford(float x) {
    unsigned u = __float_as_uint(x);
    return (u & 0x80000000u) ? ~u : (u | 0x80000000u);
}
__device__ __forceinline__ float fback(unsigned u) {
    u = (u & 0x80000000u) ? (u & 0x7FFFFFFFu) : ~u;
    return __uint_as_float(u);
}

// ---------------------------------------------------------------------------
// 1) Node GEMM: C[M,Ncols] = A[M,K] @ B[K,Ncols], f16 WMMA, f32 accum.
//    One wave per 16x16 output tile.
// ---------------------------------------------------------------------------
__global__ void gemm_node_wmma(const float* __restrict__ A,
                               const float* __restrict__ B,
                               float* __restrict__ C,
                               int M, int K, int Ncols, int totalTiles) {
    int lane   = threadIdx.x & 31;
    int wave   = threadIdx.x >> 5;
    int tileId = blockIdx.x * (blockDim.x >> 5) + wave;
    if (tileId >= totalTiles) return;

    int ntiles = Ncols >> 4;
    int mbase  = (tileId / ntiles) << 4;
    int nbase  = (tileId % ntiles) << 4;
    int r      = lane & 15;

    v8f c = {};
    for (int k0 = 0; k0 < K; k0 += 32) {
        v16h a, b;
#pragma unroll
        for (int i = 0; i < 16; ++i) {
            int kk = k0 + frag_k(lane, i);
            a[i] = (_Float16)A[(size_t)(mbase + r) * K + kk];
            b[i] = (_Float16)B[(size_t)kk * Ncols + (nbase + r)];
        }
        c = __builtin_amdgcn_wmma_f32_16x16x32_f16(false, a, false, b,
                                                   (short)0, c, false, false);
    }
    int mofs = (lane & 16) ? 8 : 0;
#pragma unroll
    for (int j = 0; j < 8; ++j)
        C[(size_t)(mbase + mofs + j) * Ncols + (nbase + r)] = c[j];
}

// ---------------------------------------------------------------------------
// 2) Edge projection score: pscore[e,h] += sum_d relu((EA@We)[e, h*dh+d])*a_p[h,d]
//    WMMA with K=16 zero-padded to 32; in-wave reduction over columns.
//    pscore must be pre-zeroed.
// ---------------------------------------------------------------------------
__global__ void edge_pscore_wmma(const float* __restrict__ EA,   // [E, 16]
                                 const float* __restrict__ We,   // [16, HD]
                                 const float* __restrict__ att,  // [H, 3*dh]
                                 float* __restrict__ pscore,     // [E, H]
                                 int HD, int dh, int H, int totalTiles) {
    int lane   = threadIdx.x & 31;
    int wave   = threadIdx.x >> 5;
    int tileId = blockIdx.x * (blockDim.x >> 5) + wave;
    if (tileId >= totalTiles) return;

    int ntiles = HD >> 4;
    int ebase  = (tileId / ntiles) << 4;
    int nbase  = (tileId % ntiles) << 4;
    int r      = lane & 15;
    int h      = nbase / dh;                    // dh==64, chunks never straddle heads
    float ap   = att[(size_t)h * 3 * dh + 2 * dh + (nbase % dh) + r];

    v16h a, b;
#pragma unroll
    for (int i = 0; i < 16; ++i) {
        int kk = frag_k(lane, i);
        a[i] = (kk < E_DIM) ? (_Float16)EA[(size_t)(ebase + r) * E_DIM + kk]
                            : (_Float16)0.0f;
        b[i] = (kk < E_DIM) ? (_Float16)We[(size_t)kk * HD + (nbase + r)]
                            : (_Float16)0.0f;
    }
    v8f c = {};
    c = __builtin_amdgcn_wmma_f32_16x16x32_f16(false, a, false, b,
                                               (short)0, c, false, false);
    float acc[8];
#pragma unroll
    for (int j = 0; j < 8; ++j)
        acc[j] = fmaxf(c[j], 0.0f) * ap;        // relu then scale by a_p[col]
    // reduce across the 16 lanes of each half-wave (columns of the tile)
#pragma unroll
    for (int m = 1; m < 16; m <<= 1)
#pragma unroll
        for (int j = 0; j < 8; ++j)
            acc[j] += __shfl_xor(acc[j], m, 32);
    if ((lane & 15) == 0) {
        int rowofs = (lane & 16) ? 8 : 0;
#pragma unroll
        for (int j = 0; j < 8; ++j)
            atomicAdd(&pscore[(size_t)(ebase + rowofs + j) * H + h], acc[j]);
    }
}

// ---------------------------------------------------------------------------
// 3) Per-node attention dots: sj[n,h]=<xh,a_j>, si[n,h]=<xh,a_i>
// ---------------------------------------------------------------------------
__global__ void node_scores(const float* __restrict__ xh,
                            const float* __restrict__ att,
                            float* __restrict__ sj, float* __restrict__ si,
                            int H, int dh) {
    int idx = blockIdx.x * blockDim.x + threadIdx.x;
    if (idx >= NN * H) return;
    int n = idx / H, h = idx % H;
    const float* aj = att + (size_t)h * 3 * dh;
    const float* ai = aj + dh;
    const float* v  = xh + (size_t)n * H * dh + (size_t)h * dh;
    float s1 = 0.f, s2 = 0.f;
    for (int d = 0; d < dh; ++d) { s1 += v[d] * aj[d]; s2 += v[d] * ai[d]; }
    sj[idx] = s1; si[idx] = s2;
}

// ---------------------------------------------------------------------------
// 4) Edge logits (leaky relu) + segment max via ordered-uint atomicMax
// ---------------------------------------------------------------------------
__global__ void edge_logit_max(const int* __restrict__ src,
                               const int* __restrict__ dst,
                               const float* __restrict__ sj,
                               const float* __restrict__ si,
                               const float* __restrict__ pscore,
                               float* __restrict__ logit,
                               unsigned* __restrict__ mmax, int H) {
    int idx = blockIdx.x * blockDim.x + threadIdx.x;
    if (idx >= EE * H) return;
    int e = idx / H, h = idx % H;
    int s = src[e], d = dst[e];
    float x = sj[(size_t)s * H + h] + si[(size_t)d * H + h] + pscore[idx];
    x = (x > 0.f) ? x : 0.2f * x;
    logit[idx] = x;
    atomicMax(&mmax[(size_t)d * H + h], ford(x));
}

// ---------------------------------------------------------------------------
// 5) exp(logit - m[dst]) -> ex (in place), segment sum via atomicAdd f32
// ---------------------------------------------------------------------------
__global__ void edge_expsum(const int* __restrict__ dst,
                            float* __restrict__ logit_ex,
                            const unsigned* __restrict__ mmax,
                            float* __restrict__ ssum, int H) {
    int idx = blockIdx.x * blockDim.x + threadIdx.x;
    if (idx >= EE * H) return;
    int e = idx / H, h = idx % H;
    int d = dst[e];
    float m  = fback(mmax[(size_t)d * H + h]);
    float ex = __expf(logit_ex[idx] - m);
    logit_ex[idx] = ex;
    atomicAdd(&ssum[(size_t)d * H + h], ex);
}

// ---------------------------------------------------------------------------
// 6) Aggregate: out[dst] += xh[src] * alpha   (16 dims per thread, f32 atomics)
// ---------------------------------------------------------------------------
__global__ void aggregate(const int* __restrict__ src,
                          const int* __restrict__ dst,
                          const float* __restrict__ ex,
                          const float* __restrict__ ssum,
                          const float* __restrict__ xh,
                          float* __restrict__ out,
                          int H, int dh, long long total) {
    long long idx = (long long)blockIdx.x * blockDim.x + threadIdx.x;
    if (idx >= total) return;
    int HD     = H * dh;
    int chunks = HD >> 4;                      // 16 dims per thread
    int e      = (int)(idx / chunks);
    int col0   = (int)(idx % chunks) << 4;
    int h      = col0 / dh;
    int s = src[e], d = dst[e];
    float alpha = ex[(size_t)e * H + h] /
                  (ssum[(size_t)d * H + h] + 1e-16f);
    const float4* xv = (const float4*)(xh + (size_t)s * HD + col0);
    float* op = out + (size_t)d * HD + col0;
#pragma unroll
    for (int q = 0; q < 4; ++q) {
        float4 v = xv[q];
        atomicAdd(op + 4 * q + 0, v.x * alpha);
        atomicAdd(op + 4 * q + 1, v.y * alpha);
        atomicAdd(op + 4 * q + 2, v.z * alpha);
        atomicAdd(op + 4 * q + 3, v.w * alpha);
    }
}

// ---------------------------------------------------------------------------
// 7) ELU (between layers)
// ---------------------------------------------------------------------------
__global__ void elu_inplace(float* __restrict__ x, long long n) {
    long long idx = (long long)blockIdx.x * blockDim.x + threadIdx.x;
    if (idx >= n) return;
    float v = x[idx];
    x[idx] = (v > 0.f) ? v : (__expf(v) - 1.0f);
}

// ---------------------------------------------------------------------------
// Host-side layer driver
// ---------------------------------------------------------------------------
static void run_layer(const float* feat_in, int K,
                      const float* W, const float* We, const float* att,
                      int H, int dh,
                      const int* src, const int* dst, const float* edge_attr,
                      float* xh, float* pscore, float* logit,
                      float* sj, float* si, unsigned* mmax, float* ssum,
                      float* out, hipStream_t stream) {
    int HD = H * dh;
    hipMemsetAsync(pscore, 0, (size_t)EE * H * sizeof(float), stream);
    hipMemsetAsync(ssum,   0, (size_t)NN * H * sizeof(float), stream);
    hipMemsetAsync(mmax,   0, (size_t)NN * H * sizeof(unsigned), stream);
    hipMemsetAsync(out,    0, (size_t)NN * HD * sizeof(float), stream);

    // node GEMM
    {
        int tiles  = (NN / 16) * (HD / 16);
        int blocks = (tiles + 7) / 8;                   // 8 waves / 256-thread block
        gemm_node_wmma<<<blocks, 256, 0, stream>>>(feat_in, W, xh, NN, K, HD, tiles);
    }
    // edge projection score
    {
        int tiles  = (EE / 16) * (HD / 16);
        int blocks = (tiles + 7) / 8;
        edge_pscore_wmma<<<blocks, 256, 0, stream>>>(edge_attr, We, att, pscore,
                                                     HD, dh, H, tiles);
    }
    // node attention dots
    {
        int n = NN * H;
        node_scores<<<(n + 255) / 256, 256, 0, stream>>>(xh, att, sj, si, H, dh);
    }
    // logits + segment max
    {
        int n = EE * H;
        edge_logit_max<<<(n + 255) / 256, 256, 0, stream>>>(src, dst, sj, si,
                                                            pscore, logit, mmax, H);
    }
    // exp + segment sum
    {
        int n = EE * H;
        edge_expsum<<<(n + 255) / 256, 256, 0, stream>>>(dst, logit, mmax, ssum, H);
    }
    // scatter aggregate
    {
        long long total = (long long)EE * (HD / 16);
        int blocks = (int)((total + 255) / 256);
        aggregate<<<blocks, 256, 0, stream>>>(src, dst, logit, ssum, xh, out,
                                              H, dh, total);
    }
}

extern "C" void kernel_launch(void* const* d_in, const int* in_sizes, int n_in,
                              void* d_out, int out_size, void* d_ws, size_t ws_size,
                              hipStream_t stream) {
    const float* x    = (const float*)d_in[0];
    const int*   ei   = (const int*)  d_in[1];   // [2, E] int32
    const float* ea   = (const float*)d_in[2];
    const float* W0   = (const float*)d_in[3];
    const float* We0  = (const float*)d_in[4];
    const float* att0 = (const float*)d_in[5];
    const float* W1   = (const float*)d_in[6];
    const float* We1  = (const float*)d_in[7];
    const float* att1 = (const float*)d_in[8];
    const float* W2   = (const float*)d_in[9];
    const float* We2  = (const float*)d_in[10];
    const float* att2 = (const float*)d_in[11];
    const int* src = ei;
    const int* dst = ei + EE;
    float* outF = (float*)d_out;                 // [N, 4*DOUT]

    // workspace carve-up (bytes)
    char* p = (char*)d_ws;
    float*    xh     = (float*)p;    p += (size_t)NN * 256 * sizeof(float);
    float*    hA     = (float*)p;    p += (size_t)NN * HIDN * sizeof(float);
    float*    hB     = (float*)p;    p += (size_t)NN * HIDN * sizeof(float);
    float*    pscore = (float*)p;    p += (size_t)EE * 4 * sizeof(float);
    float*    logit  = (float*)p;    p += (size_t)EE * 4 * sizeof(float);
    float*    sj     = (float*)p;    p += (size_t)NN * 4 * sizeof(float);
    float*    si     = (float*)p;    p += (size_t)NN * 4 * sizeof(float);
    unsigned* mmax   = (unsigned*)p; p += (size_t)NN * 4 * sizeof(unsigned);
    float*    ssum   = (float*)p;    p += (size_t)NN * 4 * sizeof(float);

    // Layer 0: [N,128] -> [N,64], 1 head
    run_layer(x, D_IN, W0, We0, att0, 1, HIDN, src, dst, ea,
              xh, pscore, logit, sj, si, mmax, ssum, hA, stream);
    elu_inplace<<<((long long)NN * HIDN + 255) / 256, 256, 0, stream>>>(
        hA, (long long)NN * HIDN);

    // Layer 1: [N,64] -> [N,64], 1 head
    run_layer(hA, HIDN, W1, We1, att1, 1, HIDN, src, dst, ea,
              xh, pscore, logit, sj, si, mmax, ssum, hB, stream);
    elu_inplace<<<((long long)NN * HIDN + 255) / 256, 256, 0, stream>>>(
        hB, (long long)NN * HIDN);

    // Layer 2: [N,64] -> [N,256], 4 heads, no ELU, write d_out
    run_layer(hB, HIDN, W2, We2, att2, 4, DOUT, src, dst, ea,
              xh, pscore, logit, sj, si, mmax, ssum, outF, stream);
}
